// PotentialNetLayer_56530359550048
// MI455X (gfx1250) — matrix operand order repaired and checked by reading, not compiled, not run on hardware
//
#include <hip/hip_runtime.h>

#define HD   128   // hidden dim
#define AD   16    // edge_attr dim
#define TT   4     // edge types
#define KIN  144   // HD + AD
#define KPAD 160   // padded to multiple of 32

typedef __attribute__((ext_vector_type(16))) __bf16 v16bf;
typedef __attribute__((ext_vector_type(8)))  float  v8f;

// packed-fragment buffer sizes (in dwords)
#define SZ1T (5 * 8 * 32 * 8)        // W1 per type: 5 K-chunks, 8 N-tiles = 40KB
#define SZ2T (4 * 8 * 32 * 8)        // W2 per type: 4 K-chunks, 8 N-tiles = 32KB
#define SZ1  (TT * SZ1T)
#define SZ2  (TT * SZ2T)
#define SZG  (4 * 24 * 32 * 8)       // w_ih^T / w_hh^T: 4 K-chunks, 24 N-tiles

union Frag {
    unsigned int u[8];
    uint4        q[2];
    v16bf        v;
};

__device__ __forceinline__ unsigned short f2bf(float x) {
    unsigned int u = __float_as_uint(x);
    unsigned int rnd = 0x7FFFu + ((u >> 16) & 1u);   // round-to-nearest-even
    return (unsigned short)((u + rnd) >> 16);
}
__device__ __forceinline__ unsigned int pack2bf(float a, float b) {
    return (unsigned int)f2bf(a) | ((unsigned int)f2bf(b) << 16);
}

// ---------------------------------------------------------------------------
// Repack f32 weights into bf16 WMMA B-fragment layout.
// B-frag (32x16 bf16, wave32), k order chosen here to match the A loads: lane L -> N = L&15 ; reg r packs
// K = 16*(L>>4) + 2r + {0,1}. Stored so a wave reads a fragment as
// 8 consecutive dwords per lane (two b128 loads, global or LDS).
// ---------------------------------------------------------------------------
__global__ void pack_weights(const float* __restrict__ W1, const float* __restrict__ W2,
                             const float* __restrict__ wih, const float* __restrict__ whh,
                             unsigned int* __restrict__ pw1, unsigned int* __restrict__ pw2,
                             unsigned int* __restrict__ pih, unsigned int* __restrict__ phh) {
    int id = blockIdx.x * blockDim.x + threadIdx.x;
    if (id < SZ1) {
        int r = id % 8, q = id / 8;
        int lane = q % 32; q /= 32;
        int nt = q % 8;    q /= 8;
        int kc = q % 5;
        int t  = q / 5;
        int n  = nt * 16 + (lane & 15);
        int k0 = kc * 32 + 16 * (lane >> 4) + 2 * r;
        float v0 = (k0     < KIN) ? W1[((size_t)t * KIN + k0)     * HD + n] : 0.f;
        float v1 = (k0 + 1 < KIN) ? W1[((size_t)t * KIN + k0 + 1) * HD + n] : 0.f;
        pw1[id] = pack2bf(v0, v1);
        return;
    }
    id -= SZ1;
    if (id < SZ2) {
        int r = id % 8, q = id / 8;
        int lane = q % 32; q /= 32;
        int nt = q % 8;    q /= 8;
        int kc = q % 4;
        int t  = q / 4;
        int n  = nt * 16 + (lane & 15);
        int k0 = kc * 32 + 16 * (lane >> 4) + 2 * r;
        pw2[id] = pack2bf(W2[((size_t)t * HD + k0)     * HD + n],
                          W2[((size_t)t * HD + k0 + 1) * HD + n]);
        return;
    }
    id -= SZ2;
    if (id < 2 * SZG) {
        const float* w = (id < SZG) ? wih : whh;
        unsigned int* p = (id < SZG) ? pih : phh;
        int id3 = (id < SZG) ? id : id - SZG;
        int r = id3 % 8, q = id3 / 8;
        int lane = q % 32; q /= 32;
        int nt = q % 24;
        int kc = q / 24;
        int n  = nt * 16 + (lane & 15);          // 0..383 (row of w_ih)
        int k0 = kc * 32 + 16 * (lane >> 4) + 2 * r;
        // B = w^T : B[k][n] = w[n][k]
        p[id3] = pack2bf(w[(size_t)n * HD + k0], w[(size_t)n * HD + k0 + 1]);
    }
}

__global__ void zero_f32(float* __restrict__ p, size_t n) {
    size_t i = (size_t)blockIdx.x * blockDim.x + threadIdx.x;
    size_t stride = (size_t)gridDim.x * blockDim.x;
    for (; i < n; i += stride) p[i] = 0.f;
}

// meta layout (ints): [0..3]=cnt  [4..7]=off  [8..11]=cursor
__global__ void zero_meta(int* __restrict__ meta) {
    if (threadIdx.x < 12) meta[threadIdx.x] = 0;
}

__global__ void count_types(const int* __restrict__ etype, int* __restrict__ cnt, int E) {
    __shared__ int lc[TT];
    if (threadIdx.x < TT) lc[threadIdx.x] = 0;
    __syncthreads();
    int i = blockIdx.x * blockDim.x + threadIdx.x;
    int stride = gridDim.x * blockDim.x;
    for (; i < E; i += stride) atomicAdd(&lc[etype[i]], 1);
    __syncthreads();
    if (threadIdx.x < TT) atomicAdd(&cnt[threadIdx.x], lc[threadIdx.x]);
}

__global__ void compute_offsets(int* __restrict__ meta) {
    if (threadIdx.x == 0 && blockIdx.x == 0) {
        int acc = 0;
        for (int t = 0; t < TT; ++t) { meta[4 + t] = acc; meta[8 + t] = acc; acc += meta[t]; }
    }
}

// Wave-aggregated bucket scatter: one atomic per (wave, type) instead of per edge.
__global__ void bin_edges(const int* __restrict__ etype, int* __restrict__ cur,
                          int* __restrict__ bucket, int E) {
    const int lane = threadIdx.x & 31;
    int i = blockIdx.x * blockDim.x + threadIdx.x;
    int stride = gridDim.x * blockDim.x;
    for (; i < ((E + 31) & ~31); i += stride) {      // keep waves converged
        bool act = i < E;
        int t = act ? etype[i] : -1;
        for (int tt = 0; tt < TT; ++tt) {
            unsigned long long m = __ballot(t == tt);
            if (m == 0ull) continue;
            int leader = __ffsll((long long)m) - 1;
            int cw = __popcll(m);
            int basepos = 0;
            if (lane == leader) basepos = atomicAdd(&cur[tt], cw);
            basepos = __shfl(basepos, leader);
            if (t == tt) {
                int rank = __popcll(m & ((1ull << lane) - 1ull));
                bucket[basepos + rank] = i;
            }
        }
    }
}

// ---------------------------------------------------------------------------
// Per-edge 2-layer MLP (bf16 WMMA, f32 accum) + scatter-add, type-binned.
// Block = 128 threads (4 waves). For each edge type the block stages that
// type's packed bf16 W1+W2 fragments in LDS (72KB, CDNA5 WGP LDS) and then
// grid-strides over 16-edge tiles of that type; one wave per tile.
// K-reductions use TWO independent WMMA accumulator chains to avoid the
// WMMA->WMMA RAW hazard NOPs (ISA 7.12.1).
// ---------------------------------------------------------------------------
__global__ void __launch_bounds__(128)
edge_mlp_binned(const float* __restrict__ h, const int* __restrict__ eidx,
                const float* __restrict__ eattr,
                const float* __restrict__ b1, const float* __restrict__ b2,
                const unsigned int* __restrict__ pw1, const unsigned int* __restrict__ pw2,
                const int* __restrict__ bucket, const int* __restrict__ meta,
                float* __restrict__ msg, int E) {
    __shared__ unsigned int   LW1[SZ1T];        // 40KB: layer-1 B fragments, one type
    __shared__ unsigned int   LW2[SZ2T];        // 32KB: layer-2 B fragments, one type
    __shared__ unsigned short A1[4][16][KPAD];  // 20KB: h_in tiles (bf16)
    __shared__ unsigned short A2[4][16][HD];    // 16KB: relu(layer1) tiles (bf16)
    const int wave = threadIdx.x >> 5;
    const int lane = threadIdx.x & 31;
    const int hi   = lane >> 4;
    const int nl   = lane & 15;

    for (int t = 0; t < TT; ++t) {
        __syncthreads();                        // previous type's reads done
        for (int i = threadIdx.x; i < SZ1T; i += 128) LW1[i] = pw1[(size_t)t * SZ1T + i];
        for (int i = threadIdx.x; i < SZ2T; i += 128) LW2[i] = pw2[(size_t)t * SZ2T + i];
        __syncthreads();                        // weights visible to all waves

        const int cT = meta[t];                 // edges of this type
        const int oT = meta[4 + t];             // bucket offset
        const int lim = oT + cT;
        const long long ntl = ((long long)cT + 15) >> 4;

        for (long long tile = (long long)blockIdx.x * 4 + wave; tile < ntl;
             tile += (long long)gridDim.x * 4) {
            const int base = oT + (int)(tile * 16);

            // ---- gather h[src] rows (one float4 per lane per row) -> bf16 LDS
            for (int m = 0; m < 16; ++m) {
                int pos = base + m;
                int e = bucket[(pos < lim) ? pos : base];
                int s = eidx[e];                // edge_index[0][e]
                float4 v = reinterpret_cast<const float4*>(h + (size_t)s * HD)[lane];
                unsigned short* row = &A1[wave][m][lane * 4];
                row[0] = f2bf(v.x); row[1] = f2bf(v.y);
                row[2] = f2bf(v.z); row[3] = f2bf(v.w);
            }
            // ---- edge_attr (16 floats per row = 4 float4)
            for (int it = 0; it < 2; ++it) {
                int idx = it * 32 + lane; int m = idx >> 2; int c = idx & 3;
                int pos = base + m;
                int e = bucket[(pos < lim) ? pos : base];
                float4 v = reinterpret_cast<const float4*>(eattr)[(size_t)e * 4 + c];
                unsigned short* row = &A1[wave][m][HD + c * 4];
                row[0] = f2bf(v.x); row[1] = f2bf(v.y);
                row[2] = f2bf(v.z); row[3] = f2bf(v.w);
            }
            // ---- zero-pad K = 144..159
            for (int it = 0; it < 4; ++it) {
                int idx = it * 32 + lane; int m = idx >> 3; int dw = idx & 7;
                *reinterpret_cast<unsigned int*>(&A1[wave][m][KIN + dw * 2]) = 0u;
            }

            // ---- per-lane metadata for owned C rows (M = j + 8*hi)
            int dstv[8]; bool vldv[8];
#pragma unroll
            for (int j = 0; j < 8; ++j) {
                int pos = base + j + 8 * hi;
                bool vld = pos < lim;
                int e = bucket[vld ? pos : base];
                vldv[j] = vld;
                dstv[j] = eidx[(size_t)E + e];  // edge_index[1][e]
            }

            // ---- A1 fragments (this code's k order, same as B: lane M=L&15; reg r: K=2r+16*(L>>4))
            Frag a1[5];
#pragma unroll
            for (int kc = 0; kc < 5; ++kc)
#pragma unroll
                for (int r = 0; r < 8; ++r) {
                    int kl = 16 * hi + 2 * r;
                    a1[kc].u[r] = *reinterpret_cast<const unsigned int*>(
                        &A1[wave][nl][kc * 32 + kl]);
                }

            // ---- layer 1: [16x160] x [160x128], B from LDS, dual accum chains
            for (int nt = 0; nt < 8; ++nt) {
                v8f c0 = {0.f, 0.f, 0.f, 0.f, 0.f, 0.f, 0.f, 0.f};
                v8f c1 = {0.f, 0.f, 0.f, 0.f, 0.f, 0.f, 0.f, 0.f};
                Frag b[5];
#pragma unroll
                for (int kc = 0; kc < 5; ++kc) {
                    const uint4* bp = reinterpret_cast<const uint4*>(
                        &LW1[(((kc * 8) + nt) * 32 + lane) * 8]);
                    b[kc].q[0] = bp[0]; b[kc].q[1] = bp[1];
                }
                // chain 0: kc = 0,2,4 ; chain 1: kc = 1,3  (independent -> no RAW NOPs)
                c0 = __builtin_amdgcn_wmma_f32_16x16x32_bf16(false, a1[0].v, false, b[0].v, (short)0, c0, false, false);
                c1 = __builtin_amdgcn_wmma_f32_16x16x32_bf16(false, a1[1].v, false, b[1].v, (short)0, c1, false, false);
                c0 = __builtin_amdgcn_wmma_f32_16x16x32_bf16(false, a1[2].v, false, b[2].v, (short)0, c0, false, false);
                c1 = __builtin_amdgcn_wmma_f32_16x16x32_bf16(false, a1[3].v, false, b[3].v, (short)0, c1, false, false);
                c0 = __builtin_amdgcn_wmma_f32_16x16x32_bf16(false, a1[4].v, false, b[4].v, (short)0, c0, false, false);
                float bias = b1[t * HD + nt * 16 + nl];
#pragma unroll
                for (int j = 0; j < 8; ++j) {
                    float f = c0[j] + c1[j] + bias;
                    f = f > 0.f ? f : 0.f;      // ReLU
                    A2[wave][j + 8 * hi][nt * 16 + nl] = f2bf(f);
                }
            }
            // ---- A2 fragments
            Frag a2[4];
#pragma unroll
            for (int kc = 0; kc < 4; ++kc)
#pragma unroll
                for (int r = 0; r < 8; ++r) {
                    int kl = 16 * hi + 2 * r;
                    a2[kc].u[r] = *reinterpret_cast<const unsigned int*>(
                        &A2[wave][nl][kc * 32 + kl]);
                }
            // ---- layer 2 + scatter-add, dual accum chains
            for (int nt = 0; nt < 8; ++nt) {
                v8f c0 = {0.f, 0.f, 0.f, 0.f, 0.f, 0.f, 0.f, 0.f};
                v8f c1 = {0.f, 0.f, 0.f, 0.f, 0.f, 0.f, 0.f, 0.f};
                Frag b[4];
#pragma unroll
                for (int kc = 0; kc < 4; ++kc) {
                    const uint4* bp = reinterpret_cast<const uint4*>(
                        &LW2[(((kc * 8) + nt) * 32 + lane) * 8]);
                    b[kc].q[0] = bp[0]; b[kc].q[1] = bp[1];
                }
                c0 = __builtin_amdgcn_wmma_f32_16x16x32_bf16(false, a2[0].v, false, b[0].v, (short)0, c0, false, false);
                c1 = __builtin_amdgcn_wmma_f32_16x16x32_bf16(false, a2[1].v, false, b[1].v, (short)0, c1, false, false);
                c0 = __builtin_amdgcn_wmma_f32_16x16x32_bf16(false, a2[2].v, false, b[2].v, (short)0, c0, false, false);
                c1 = __builtin_amdgcn_wmma_f32_16x16x32_bf16(false, a2[3].v, false, b[3].v, (short)0, c1, false, false);
                float bias2 = b2[t * HD + nt * 16 + nl];
#pragma unroll
                for (int j = 0; j < 8; ++j) {
                    if (vldv[j])
                        atomicAdd(&msg[(size_t)dstv[j] * HD + nt * 16 + nl],
                                  c0[j] + c1[j] + bias2);
                }
            }
        }
    }
}

// ---------------------------------------------------------------------------
// GRU cell: gi = msg @ w_ih^T + b_ih ; gh = h @ w_hh^T + b_hh ; gates r,z,n.
// One wave per tile of 16 nodes; 6 independent WMMA accumulator chains.
// ---------------------------------------------------------------------------
__global__ void __launch_bounds__(128)
gru_update(const float* __restrict__ msg, const float* __restrict__ h,
           const float* __restrict__ bih, const float* __restrict__ bhh,
           const unsigned int* __restrict__ pih, const unsigned int* __restrict__ phh,
           float* __restrict__ out, int N) {
    __shared__ unsigned short SM[4][16][HD];
    __shared__ unsigned short SH[4][16][HD];
    const int wave = threadIdx.x >> 5;
    const int lane = threadIdx.x & 31;
    const int hi   = lane >> 4;
    const int nl   = lane & 15;
    const long long n0 = ((long long)blockIdx.x * 4 + wave) * 16;
    if (n0 >= N) return;                        // wave-uniform

    for (int m = 0; m < 16; ++m) {
        long long row = n0 + m; if (row >= N) row = N - 1;
        float4 vm = reinterpret_cast<const float4*>(msg + (size_t)row * HD)[lane];
        float4 vh = reinterpret_cast<const float4*>(h   + (size_t)row * HD)[lane];
        unsigned short* rm = &SM[wave][m][lane * 4];
        unsigned short* rh = &SH[wave][m][lane * 4];
        rm[0] = f2bf(vm.x); rm[1] = f2bf(vm.y); rm[2] = f2bf(vm.z); rm[3] = f2bf(vm.w);
        rh[0] = f2bf(vh.x); rh[1] = f2bf(vh.y); rh[2] = f2bf(vh.z); rh[3] = f2bf(vh.w);
    }

    Frag am[4], ah[4];
#pragma unroll
    for (int kc = 0; kc < 4; ++kc)
#pragma unroll
        for (int r = 0; r < 8; ++r) {
            int kl = 16 * hi + 2 * r;
            am[kc].u[r] = *reinterpret_cast<const unsigned int*>(&SM[wave][nl][kc * 32 + kl]);
            ah[kc].u[r] = *reinterpret_cast<const unsigned int*>(&SH[wave][nl][kc * 32 + kl]);
        }

    for (int ng = 0; ng < 8; ++ng) {            // column tile within a gate
        v8f cir = {0,0,0,0,0,0,0,0}, chr_ = {0,0,0,0,0,0,0,0};
        v8f ciz = {0,0,0,0,0,0,0,0}, chz  = {0,0,0,0,0,0,0,0};
        v8f cin = {0,0,0,0,0,0,0,0}, chn  = {0,0,0,0,0,0,0,0};
#pragma unroll
        for (int kc = 0; kc < 4; ++kc) {
#pragma unroll
            for (int g = 0; g < 3; ++g) {       // gate: r, z, n
                int nt = ng + 8 * g;
                Frag bi, bh;
                const uint4* pi = reinterpret_cast<const uint4*>(
                    pih + (((size_t)kc * 24 + nt) * 32 + lane) * 8);
                const uint4* ph = reinterpret_cast<const uint4*>(
                    phh + (((size_t)kc * 24 + nt) * 32 + lane) * 8);
                bi.q[0] = pi[0]; bi.q[1] = pi[1];
                bh.q[0] = ph[0]; bh.q[1] = ph[1];
                if (g == 0) {
                    cir  = __builtin_amdgcn_wmma_f32_16x16x32_bf16(false, am[kc].v, false, bi.v, (short)0, cir,  false, false);
                    chr_ = __builtin_amdgcn_wmma_f32_16x16x32_bf16(false, ah[kc].v, false, bh.v, (short)0, chr_, false, false);
                } else if (g == 1) {
                    ciz = __builtin_amdgcn_wmma_f32_16x16x32_bf16(false, am[kc].v, false, bi.v, (short)0, ciz, false, false);
                    chz = __builtin_amdgcn_wmma_f32_16x16x32_bf16(false, ah[kc].v, false, bh.v, (short)0, chz, false, false);
                } else {
                    cin = __builtin_amdgcn_wmma_f32_16x16x32_bf16(false, am[kc].v, false, bi.v, (short)0, cin, false, false);
                    chn = __builtin_amdgcn_wmma_f32_16x16x32_bf16(false, ah[kc].v, false, bh.v, (short)0, chn, false, false);
                }
            }
        }
        int col = ng * 16 + nl;
        float bir = bih[col],          bhr = bhh[col];
        float biz = bih[HD + col],     bhz = bhh[HD + col];
        float bin = bih[2 * HD + col], bhn = bhh[2 * HD + col];
#pragma unroll
        for (int j = 0; j < 8; ++j) {
            long long node = n0 + j + 8 * hi;
            float r  = 1.f / (1.f + __expf(-((cir[j] + bir) + (chr_[j] + bhr))));
            float z  = 1.f / (1.f + __expf(-((ciz[j] + biz) + (chz[j]  + bhz))));
            float nn = tanhf((cin[j] + bin) + r * (chn[j] + bhn));
            if (node < N) {
                float hold = h[(size_t)node * HD + col];
                out[(size_t)node * HD + col] = (1.f - z) * nn + z * hold;
            }
        }
    }
}

extern "C" void kernel_launch(void* const* d_in, const int* in_sizes, int n_in,
                              void* d_out, int out_size, void* d_ws, size_t ws_size,
                              hipStream_t stream) {
    const float* h     = (const float*)d_in[0];
    const int*   eidx  = (const int*)  d_in[1];
    const int*   etype = (const int*)  d_in[2];
    const float* eattr = (const float*)d_in[3];
    const float* W1    = (const float*)d_in[4];
    const float* b1    = (const float*)d_in[5];
    const float* W2    = (const float*)d_in[6];
    const float* b2    = (const float*)d_in[7];
    const float* wih   = (const float*)d_in[8];
    const float* whh   = (const float*)d_in[9];
    const float* bih   = (const float*)d_in[10];
    const float* bhh   = (const float*)d_in[11];

    const int N = in_sizes[0] / HD;
    const int E = in_sizes[2];

    // workspace: [ msg f32 N*HD ][ pw1 ][ pw2 ][ pih ][ phh ][ meta 16 ][ bucket E ]
    float*        msg  = (float*)d_ws;
    unsigned int* pw1  = (unsigned int*)d_ws + (size_t)N * HD;
    unsigned int* pw2  = pw1 + SZ1;
    unsigned int* pih  = pw2 + SZ2;
    unsigned int* phh  = pih + SZG;
    int*          meta = (int*)(phh + SZG);
    int*          bucket = meta + 16;

    const int packTotal = SZ1 + SZ2 + 2 * SZG;
    pack_weights<<<(packTotal + 255) / 256, 256, 0, stream>>>(W1, W2, wih, whh,
                                                              pw1, pw2, pih, phh);
    zero_f32<<<1024, 256, 0, stream>>>(msg, (size_t)N * HD);
    zero_meta<<<1, 32, 0, stream>>>(meta);
    count_types<<<1024, 256, 0, stream>>>(etype, meta, E);
    compute_offsets<<<1, 32, 0, stream>>>(meta);
    bin_edges<<<1024, 256, 0, stream>>>(etype, meta + 8, bucket, E);

    edge_mlp_binned<<<1280, 128, 0, stream>>>(h, eidx, eattr, b1, b2,
                                              pw1, pw2, bucket, meta, msg, E);

    long long ntiles = ((long long)N + 15) / 16;
    gru_update<<<(unsigned)((ntiles + 3) / 4), 128, 0, stream>>>(
        msg, h, bih, bhh, pih, phh, (float*)d_out, N);
}
